// HeteroGraphTransformer_28467043238273
// MI455X (gfx1250) — matrix operand order repaired and checked by reading, not compiled, not run on hardware
//
#include <hip/hip_runtime.h>
#include <math.h>

// ---------------------------------------------------------------------------
// HeteroGraphTransformer forward for MI455X (gfx1250, wave32).
// Dense projections use v_wmma_f32_16x16x32_bf16, 2x2 tiles per wave, operands
// staged in LDS as bf16 with fragment-contiguous layouts so each fragment is
// two ds_load_b128. Edge softmax/aggregation uses L2-resident f32 atomics
// (node buffers ~30MB << 192MB L2). Row reductions use wave32 shfl butterflies.
// ---------------------------------------------------------------------------

#define HC 128   // hidden channels
#define HH 8     // heads
#define CC 16    // channels per head
#define NLAYERS 4

typedef __attribute__((ext_vector_type(8)))  __bf16 v8bf;
typedef __attribute__((ext_vector_type(16))) __bf16 v16bf;
typedef __attribute__((ext_vector_type(8)))  float  v8f;

__device__ __forceinline__ float wave_allsum(float v) {
#pragma unroll
  for (int m = 16; m >= 1; m >>= 1) v += __shfl_xor(v, m, 32);
  return v;
}

__device__ __forceinline__ float gelu_exact(float x) {
  return 0.5f * x * (1.0f + erff(x * 0.70710678118654752f));
}

// order-preserving float<->uint for atomicMax-based segment max
__device__ __forceinline__ unsigned f32_ord(float f) {
  unsigned u = __float_as_uint(f);
  return (u & 0x80000000u) ? ~u : (u | 0x80000000u);
}
__device__ __forceinline__ float ord_f32(unsigned u) {
  unsigned b = (u & 0x80000000u) ? (u ^ 0x80000000u) : ~u;
  return __uint_as_float(b);
}

// ---------------------------------------------------------------------------
// C[M,N] = op(A[M,K] @ W[K,N] + bias) (+ D), fp32 in/out, bf16 WMMA inner
// product. Requires N % 128 == 0, K % 32 == 0 (all uses: N in {128,512},
// K in {64,128,512}). M-edge guarded at row granularity.
//
// Block: 256 threads = 8 waves as 2(M) x 4(N); each wave computes a 32x32
// output patch = 2x2 WMMA 16x16 tiles; block tile 64(M) x 128(N), K-step 32.
//
// LDS layouts (80-byte row stride keeps every fragment chunk 16B aligned):
//   As [64 rows][32 K]  row-major : A fragment = 2 x ds_load_b128 per lane
//   BsT[128 cols][32 K] transposed: B fragment = 2 x ds_load_b128 per lane
// ---------------------------------------------------------------------------
__global__ __launch_bounds__(256) void wmma_gemm_kernel(
    const float* __restrict__ A, const float* __restrict__ W,
    const float* __restrict__ bias, const float* __restrict__ D,
    float* __restrict__ C, int M, int N, int K, int act)
{
  __shared__ __bf16 As[64][40];
  __shared__ __bf16 BsT[128][40];

  const int tid  = threadIdx.x;
  const int lane = tid & 31;
  const int wid  = tid >> 5;
  const int wm   = wid >> 2;      // 0..1  (32 M rows each)
  const int wn   = wid & 3;       // 0..3  (32 N cols each)
  const int m0   = blockIdx.y * 64;
  const int n0   = blockIdx.x * 128;

  v8f acc[2][2] = {};

  // A staging: thread -> (row, 8-float segment); incremental pointer over K.
  const int ar_st = tid >> 2;        // 0..63
  const int ac_st = (tid & 3) * 8;   // 0,8,16,24
  const int gr_st = m0 + ar_st;
  const bool a_ok = (gr_st < M);
  const float* ap = &A[(size_t)gr_st * K + ac_st];

  // W staging: thread -> (col, even-K row pair), incremental pointer over K.
  const int wc_st = tid & 127;            // N col 0..127
  const int wr_st = (tid >> 7) * 2;       // 0 or 2
  const float* wp0 = &W[(size_t)wr_st * N + (n0 + wc_st)];
  const size_t wstep = 4 * (size_t)N;     // 4 K rows per staging iteration

  for (int k0 = 0; k0 < K; k0 += 32) {
    // ---- stage A tile (64 x 32) as bf16, one ds_store_b128 per thread ----
    {
      v8bf pk = {};
      if (a_ok) {
        const float4* a4 = reinterpret_cast<const float4*>(ap);
        float4 f0 = a4[0];
        float4 f1 = a4[1];
        pk[0] = (__bf16)f0.x; pk[1] = (__bf16)f0.y;
        pk[2] = (__bf16)f0.z; pk[3] = (__bf16)f0.w;
        pk[4] = (__bf16)f1.x; pk[5] = (__bf16)f1.y;
        pk[6] = (__bf16)f1.z; pk[7] = (__bf16)f1.w;
      }
      *reinterpret_cast<v8bf*>(&As[ar_st][ac_st]) = pk;
      // prefetch next K tile of A (global_prefetch_b8)
      if (a_ok && k0 + 32 < K) __builtin_prefetch(ap + 32, 0, 1);
      ap += 32;
    }
    // ---- stage W tile (32 x 128) transposed; K-pairs packed to u32 ----
    {
      const float* wp = wp0;
#pragma unroll
      for (int i = 0; i < 8; ++i) {
        unsigned short u0 = __builtin_bit_cast(unsigned short, (__bf16)wp[0]);
        unsigned short u1 = __builtin_bit_cast(unsigned short, (__bf16)wp[N]);
        *reinterpret_cast<unsigned*>(&BsT[wc_st][wr_st + i * 4]) =
            ((unsigned)u1 << 16) | u0;
        wp += wstep;
      }
      wp0 += 32 * (size_t)N;
    }
    __syncthreads();

    // ---- fragments ----
    // A 16-bit 16x32 layout: lane%16 = M row; lanes 0-15 hold K {0-7,16-23},
    // lanes 16-31 hold K {8-15,24-31}.
    const int ka = (lane < 16) ? 0 : 8;
    // B 16-bit 32x16 layout: lane%16 = N col; lanes 0-15 K {0-7,16-23}, 16-31 K {8-15,24-31}.
    const int kb = (lane < 16) ? 0 : 8;

    v16bf afr[2], bfr[2];
#pragma unroll
    for (int mt = 0; mt < 2; ++mt) {
      int ar = wm * 32 + mt * 16 + (lane & 15);
      v8bf lo = *reinterpret_cast<const v8bf*>(&As[ar][ka]);
      v8bf hi = *reinterpret_cast<const v8bf*>(&As[ar][ka + 16]);
      afr[mt] = __builtin_shufflevector(lo, hi, 0, 1, 2, 3, 4, 5, 6, 7,
                                        8, 9, 10, 11, 12, 13, 14, 15);
    }
#pragma unroll
    for (int nt = 0; nt < 2; ++nt) {
      int bc = wn * 32 + nt * 16 + (lane & 15);
      v8bf lo = *reinterpret_cast<const v8bf*>(&BsT[bc][kb]);
      v8bf hi = *reinterpret_cast<const v8bf*>(&BsT[bc][kb + 16]);
      bfr[nt] = __builtin_shufflevector(lo, hi, 0, 1, 2, 3, 4, 5, 6, 7,
                                        8, 9, 10, 11, 12, 13, 14, 15);
    }

#pragma unroll
    for (int mt = 0; mt < 2; ++mt)
#pragma unroll
      for (int nt = 0; nt < 2; ++nt)
        acc[mt][nt] = __builtin_amdgcn_wmma_f32_16x16x32_bf16(
            false, afr[mt], false, bfr[nt], (short)0, acc[mt][nt], false, false);
    __syncthreads();
  }

  // C layout: VGPR r -> M = r (lanes 0-15) / M = 8+r (lanes 16-31), N = lane%16
  const int coff = (lane < 16) ? 0 : 8;
#pragma unroll
  for (int mt = 0; mt < 2; ++mt) {
#pragma unroll
    for (int nt = 0; nt < 2; ++nt) {
      int col = n0 + wn * 32 + nt * 16 + (lane & 15);
#pragma unroll
      for (int r = 0; r < 8; ++r) {
        int row = m0 + wm * 32 + mt * 16 + coff + r;
        if (row < M) {
          float v = acc[mt][nt][r];
          if (bias) v += bias[col];
          if (act) v = gelu_exact(v);
          if (D) v += D[(size_t)row * N + col];
          C[(size_t)row * N + col] = v;
        }
      }
    }
  }
}

// ---------------------------------------------------------------------------
__global__ void fill_f32(float* __restrict__ p, float v, int n) {
  int i = blockIdx.x * 256 + threadIdx.x;
  if (i < n) p[i] = v;
}
__global__ void fill_u32(unsigned* __restrict__ p, unsigned v, int n) {
  int i = blockIdx.x * 256 + threadIdx.x;
  if (i < n) p[i] = v;
}

// logits[e,h] = <q[dst,h,:], k[src,h,:] + ea[e]*We[h,:]> / sqrt(C); seg-max via
// ordered-uint atomicMax into mmax[dst,h].
__global__ __launch_bounds__(256) void edge_logits_kernel(
    const float* __restrict__ q, const float* __restrict__ k,
    const float* __restrict__ ea, const float* __restrict__ We,
    const int* __restrict__ src, const int* __restrict__ dst,
    float* __restrict__ logits, unsigned* __restrict__ mmax, int E)
{
  int idx = blockIdx.x * 256 + threadIdx.x;
  if (idx >= E * HH) return;
  int e = idx >> 3, h = idx & 7;
  int s = src[e], d = dst[e];
  float a = ea[e];
  const float* qp = q + (size_t)d * HC + h * CC;
  const float* kp = k + (size_t)s * HC + h * CC;
  const float* wp = We + h * CC;
  float acc = 0.0f;
#pragma unroll
  for (int c = 0; c < CC; ++c) acc += qp[c] * (kp[c] + a * wp[c]);
  acc *= 0.25f;  // 1/sqrt(16)
  logits[idx] = acc;
  atomicMax(&mmax[d * HH + h], f32_ord(acc));
}

// decode ordered max in place (non-finite -> 0, matches reference), zero s
__global__ void seg_decode_kernel(unsigned* __restrict__ m,
                                  float* __restrict__ s, int n)
{
  int i = blockIdx.x * 256 + threadIdx.x;
  if (i >= n) return;
  float v = ord_f32(m[i]);
  ((float*)m)[i] = (v > -3.0e38f && v < 3.0e38f) ? v : 0.0f;
  s[i] = 0.0f;
}

// p = exp(logit - m[dst,h]); s[dst,h] += p; out[dst,h,:] += p*(v[src,h,:]+e)
__global__ __launch_bounds__(256) void edge_accum_kernel(
    const float* __restrict__ v, const float* __restrict__ ea,
    const float* __restrict__ We, const int* __restrict__ src,
    const int* __restrict__ dst, const float* __restrict__ logits,
    const float* __restrict__ m, float* __restrict__ ssum,
    float* __restrict__ outacc, int E)
{
  int idx = blockIdx.x * 256 + threadIdx.x;
  if (idx >= E * HH) return;
  int e = idx >> 3, h = idx & 7;
  int s = src[e], d = dst[e];
  float a = ea[e];
  float p = expf(logits[idx] - m[d * HH + h]);
  atomicAdd(&ssum[d * HH + h], p);
  const float* vp = v + (size_t)s * HC + h * CC;
  const float* wp = We + h * CC;
  float* op = outacc + (size_t)d * HC + h * CC;
#pragma unroll
  for (int c = 0; c < CC; ++c) atomicAdd(&op[c], p * (vp[c] + a * wp[c]));
}

// one wave32 per node: normalize, beta-gate, accumulate into agg
__global__ __launch_bounds__(256) void node_combine_kernel(
    const float* __restrict__ outacc, const float* __restrict__ ssum,
    const float* __restrict__ xr, const float* __restrict__ beta,
    float* __restrict__ agg, int nd)
{
  int lane = threadIdx.x & 31;
  int node = blockIdx.x * 8 + (threadIdx.x >> 5);
  if (node >= nd) return;
  float o[4], r[4], part = 0.0f;
#pragma unroll
  for (int i = 0; i < 4; ++i) {
    int c = i * 32 + lane;
    float s = ssum[node * HH + (c >> 4)] + 1e-16f;
    o[i] = outacc[(size_t)node * HC + c] / s;
    r[i] = xr[(size_t)node * HC + c];
    part += beta[c] * o[i] + beta[HC + c] * r[i] + beta[2 * HC + c] * (o[i] - r[i]);
  }
  float g = wave_allsum(part);
  g = 1.0f / (1.0f + expf(-g));
#pragma unroll
  for (int i = 0; i < 4; ++i) {
    int c = i * 32 + lane;
    agg[(size_t)node * HC + c] += g * r[i] + (1.0f - g) * o[i];
  }
}

// one wave32 per node: x = LN(agg + x) * gamma + beta (in place on x)
__global__ __launch_bounds__(256) void ln_residual_kernel(
    const float* __restrict__ agg, const float* __restrict__ gamma,
    const float* __restrict__ beta, float* __restrict__ x, int n)
{
  int lane = threadIdx.x & 31;
  int node = blockIdx.x * 8 + (threadIdx.x >> 5);
  if (node >= n) return;
  float v[4];
  float s = 0.0f;
#pragma unroll
  for (int i = 0; i < 4; ++i) {
    int c = i * 32 + lane;
    v[i] = agg[(size_t)node * HC + c] + x[(size_t)node * HC + c];
    s += v[i];
  }
  float mean = wave_allsum(s) * (1.0f / HC);
  float var = 0.0f;
#pragma unroll
  for (int i = 0; i < 4; ++i) { float d = v[i] - mean; var += d * d; }
  var = wave_allsum(var) * (1.0f / HC);
  float inv = rsqrtf(var + 1e-5f);
#pragma unroll
  for (int i = 0; i < 4; ++i) {
    int c = i * 32 + lane;
    x[(size_t)node * HC + c] = (v[i] - mean) * inv * gamma[c] + beta[c];
  }
}

// ---------------------------------------------------------------------------
extern "C" void kernel_launch(void* const* d_in, const int* in_sizes, int n_in,
                              void* d_out, int out_size, void* d_ws, size_t ws_size,
                              hipStream_t stream)
{
  (void)n_in; (void)out_size; (void)ws_size;

  // ---- inputs (setup_inputs() dict order; params flattened depth-first) ----
  const float* x_drug = (const float*)d_in[0];   // [ND, 64]
  const float* x_prot = (const float*)d_in[1];   // [NP, 128]
  const int ND = in_sizes[0] / 64;
  const int NP = in_sizes[1] / 128;
  const int* ei[3] = {(const int*)d_in[2], (const int*)d_in[4], (const int*)d_in[6]};
  const float* ea[3] = {(const float*)d_in[3], (const float*)d_in[5], (const float*)d_in[7]};
  int E[3] = {in_sizes[2] / 2, in_sizes[4] / 2, in_sizes[6] / 2};
  const int esrc_t[3] = {0, 1, 1};   // d2p, p2d, p2p (drug=0, protein=1)
  const int edst_t[3] = {1, 0, 1};
  int Ncnt[2] = {ND, NP};

  int pi = 8;
  auto P = [&]() { return (const float*)d_in[pi++]; };
  const float *ipW[2], *ipB[2], *opW[2], *opB[2];
  for (int t = 0; t < 2; ++t) { ipW[t] = P(); ipB[t] = P(); }
  for (int t = 0; t < 2; ++t) { opW[t] = P(); opB[t] = P(); }
  struct ConvP { const float *qW,*qB,*kW,*kB,*vW,*vB,*eW,*skW,*skB,*beta; };
  struct LayerP {
    ConvP cv[3];
    const float *lnG[2], *lnB[2];
    const float *f1W[2], *f1B[2], *f2W[2], *f2B[2];
  };
  LayerP LP[NLAYERS];
  for (int l = 0; l < NLAYERS; ++l) {
    for (int e = 0; e < 3; ++e) {
      ConvP& c = LP[l].cv[e];
      c.qW = P(); c.qB = P(); c.kW = P(); c.kB = P(); c.vW = P(); c.vB = P();
      c.eW = P(); c.skW = P(); c.skB = P(); c.beta = P();
    }
    for (int t = 0; t < 2; ++t) { LP[l].lnG[t] = P(); LP[l].lnB[t] = P(); }
    for (int t = 0; t < 2; ++t) {
      LP[l].f1W[t] = P(); LP[l].f1B[t] = P();
      LP[l].f2W[t] = P(); LP[l].f2B[t] = P();
    }
  }

  // ---- workspace carve-up (floats) ----
  float* w = (float*)d_ws;
  size_t off = 0;
  auto alloc = [&](size_t nf) { float* p = w + off; off += nf; return p; };
  float* x[2];   x[0]   = alloc((size_t)ND * HC); x[1]   = alloc((size_t)NP * HC);
  float* agg[2]; agg[0] = alloc((size_t)ND * HC); agg[1] = alloc((size_t)NP * HC);
  float* qb  = alloc((size_t)NP * HC);
  float* kb  = alloc((size_t)NP * HC);
  float* vb  = alloc((size_t)NP * HC);
  float* skb = alloc((size_t)NP * HC);
  float* oac = alloc((size_t)NP * HC);
  int Emax = E[0]; for (int i = 1; i < 3; ++i) if (E[i] > Emax) Emax = E[i];
  float*    logits = alloc((size_t)Emax * HH);
  float*    sbuf   = alloc((size_t)NP * HH);
  unsigned* mbuf   = (unsigned*)alloc((size_t)NP * HH);
  float*    hbuf   = alloc((size_t)NP * 4 * HC);

  // ---- helpers ----
  auto gemm = [&](const float* A, const float* W, const float* B, const float* D,
                  float* C, int M, int N, int K, int act) {
    dim3 g(N / 128, (M + 63) / 64);
    wmma_gemm_kernel<<<g, 256, 0, stream>>>(A, W, B, D, C, M, N, K, act);
  };
  auto fillf = [&](float* p, float v, size_t n) {
    fill_f32<<<(int)((n + 255) / 256), 256, 0, stream>>>(p, v, (int)n);
  };
  auto fillu = [&](unsigned* p, unsigned v, size_t n) {
    fill_u32<<<(int)((n + 255) / 256), 256, 0, stream>>>(p, v, (int)n);
  };

  // ---- in_proj ----
  gemm(x_drug, ipW[0], ipB[0], nullptr, x[0], ND, HC, 64, 0);
  gemm(x_prot, ipW[1], ipB[1], nullptr, x[1], NP, HC, HC, 0);

  // ---- layers ----
  for (int l = 0; l < NLAYERS; ++l) {
    fillf(agg[0], 0.0f, (size_t)ND * HC);
    fillf(agg[1], 0.0f, (size_t)NP * HC);

    for (int et = 0; et < 3; ++et) {
      const ConvP& c = LP[l].cv[et];
      int st = esrc_t[et], dt = edst_t[et];
      int ns = Ncnt[st], nd = Ncnt[dt];

      gemm(x[dt], c.qW,  c.qB,  nullptr, qb,  nd, HC, HC, 0);
      gemm(x[st], c.kW,  c.kB,  nullptr, kb,  ns, HC, HC, 0);
      gemm(x[st], c.vW,  c.vB,  nullptr, vb,  ns, HC, HC, 0);
      gemm(x[dt], c.skW, c.skB, nullptr, skb, nd, HC, HC, 0);

      fillu(mbuf, 0x007FFFFFu, (size_t)nd * HH);   // ordered(-inf)
      fillf(oac, 0.0f, (size_t)nd * HC);

      int eh = E[et] * HH;
      edge_logits_kernel<<<(eh + 255) / 256, 256, 0, stream>>>(
          qb, kb, ea[et], c.eW, ei[et], ei[et] + E[et], logits, mbuf, E[et]);
      seg_decode_kernel<<<((nd * HH) + 255) / 256, 256, 0, stream>>>(
          mbuf, sbuf, nd * HH);
      edge_accum_kernel<<<(eh + 255) / 256, 256, 0, stream>>>(
          vb, ea[et], c.eW, ei[et], ei[et] + E[et], logits,
          (const float*)mbuf, sbuf, oac, E[et]);
      node_combine_kernel<<<(nd + 7) / 8, 256, 0, stream>>>(
          oac, sbuf, skb, c.beta, agg[dt], nd);
    }

    for (int t = 0; t < 2; ++t) {
      ln_residual_kernel<<<(Ncnt[t] + 7) / 8, 256, 0, stream>>>(
          agg[t], LP[l].lnG[t], LP[l].lnB[t], x[t], Ncnt[t]);
      // FFN: h = gelu(x@W1+b1); x = h@W2+b2 + x
      gemm(x[t], LP[l].f1W[t], LP[l].f1B[t], nullptr, hbuf, Ncnt[t], 4 * HC, HC, 1);
      gemm(hbuf, LP[l].f2W[t], LP[l].f2B[t], x[t], x[t], Ncnt[t], HC, 4 * HC, 0);
    }
  }

  // ---- out_proj (outputs concatenated: drug then protein) ----
  float* outp = (float*)d_out;
  gemm(x[0], opW[0], opB[0], nullptr, outp, ND, HC, HC, 0);
  gemm(x[1], opW[1], opB[1], nullptr, outp + (size_t)ND * HC, NP, HC, HC, 0);
}
